// IDMForwardSim_64665027609070
// MI455X (gfx1250) — matrix operand order repaired and checked by reading, not compiled, not run on hardware
//
#include <hip/hip_runtime.h>
#include <cstdint>

typedef __attribute__((ext_vector_type(16))) _Float16 v16h;
typedef __attribute__((ext_vector_type(8)))  _Float16 v8h;
typedef __attribute__((ext_vector_type(8)))  float    v8f;

#define B_N 16384
#define T_N 60
#define ZD  6
#define H_N 100
#define HP  112     // hidden padded to 7*16
#define KP  128     // K padded to 4*32
#define NP  448     // 4 gates * HP
#define NTILES 28   // NP/16
#define WH_PACK_HALVES (NTILES*4*32*16)   // 57344 halves = 114688 B

union U8f  { v8f  v; float    f[8];  };
union U16h { v16h v; v8h      h[2];  };

__device__ __forceinline__ float sigf(float x){ return 1.0f/(1.0f + __expf(-x)); }
__device__ __forceinline__ float clipf(float x,float lo,float hi){ return fminf(fmaxf(x,lo),hi); }

__device__ __forceinline__ float idm_act(float vel,float dv,float dx,
                                         float desired_v,float tgap,float jamx,
                                         float amax,float amin){
  dx = clipf(dx, 0.5f, 1000.f);
  float gap = jamx + fmaxf(0.f, tgap*vel + vel*dv/(2.f*__fsqrt_rn(amax*amin)));
  float r = vel/desired_v; float r2 = r*r;
  float g = gap/dx;
  return clipf(amax*(1.f - r2*r2 - g*g), -3.f, 3.f);
}

__device__ __forceinline__ float hash_norm(uint32_t a, uint32_t b){
  uint32_t x = a*0x9E3779B9u ^ b*0x85EBCA6Bu ^ 0x27220A95u;
  x ^= x>>16; x *= 0x7FEB352Du; x ^= x>>15; x *= 0x846CA68Bu; x ^= x>>16;
  uint32_t y = x*0xC2B2AE35u + 0x165667B1u; y ^= y>>13; y *= 0x9E3779B1u; y ^= y>>16;
  float u1 = (float)(x>>8)*(1.f/16777216.f) + 1e-7f;
  float u2 = (float)(y>>8)*(1.f/16777216.f);
  return __fsqrt_rn(-2.f*__logf(u1))*__cosf(6.28318530718f*u2);
}

// proj = z @ W1 + b1, padded to HP cols (pad = 0)
__global__ void proj_kernel(const float* __restrict__ z, const float* __restrict__ W1,
                            const float* __restrict__ b1, float* __restrict__ projP){
  int idx = blockIdx.x*blockDim.x + threadIdx.x;
  if (idx >= B_N*HP) return;
  int h = idx % HP, b = idx / HP;
  float acc = 0.f;
  if (h < H_N){
    #pragma unroll
    for (int k=0;k<ZD;k++) acc += z[b*ZD+k]*W1[k*H_N+h];
    acc += b1[h];
  }
  projP[idx] = acc;
}

// Pack Wh (100x400 f32) into f16 WMMA B-fragment order:
// fragment f=(gate*7+jt)*4+kc, lane L, half s:  K = kc*32 + (L>>4)*16 + s,
// N-col = jt*16 + (L&15)  (contiguous k order, same as the A fragments, wave32).
__global__ void pack_wh_kernel(const float* __restrict__ Wh, _Float16* __restrict__ whPack){
  int tid = blockIdx.x*blockDim.x + threadIdx.x;
  if (tid >= WH_PACK_HALVES) return;
  int s  = tid & 15;
  int L  = (tid>>4) & 31;
  int kc = (tid>>9) & 3;
  int nt = tid >> 11;                 // 0..27
  int gate = nt / 7, jt = nt % 7;
  int j = jt*16 + (L & 15);
  int k = kc*32 + (L>>4)*16 + s;
  float v = 0.f;
  if (k < H_N && j < H_N) v = Wh[k*(4*H_N) + gate*H_N + j];
  whPack[tid] = (_Float16)v;
}

// Pack Wi's sdv row (row 100) and Wa into padded layouts (pad = 0).
__global__ void pack_vec_kernel(const float* __restrict__ Wi, const float* __restrict__ Wa,
                                float* __restrict__ wiRowP, float* __restrict__ waP){
  int tid = blockIdx.x*blockDim.x + threadIdx.x;
  if (tid < NP){
    int gate = tid/HP, j = tid%HP;
    wiRowP[tid] = (j<H_N) ? Wi[H_N*(4*H_N) + gate*H_N + j] : 0.f;
  } else if (tid < NP+HP){
    int j = tid - NP;
    waP[j] = (j<H_N) ? Wa[j] : 0.f;
  }
}

// xc = proj @ Wi[0:100,:] + bl, padded to NP cols (time-invariant LSTM input part)
__global__ void xc_kernel(const float* __restrict__ projP, const float* __restrict__ Wi,
                          const float* __restrict__ bl, float* __restrict__ xcP){
  int idx = blockIdx.x*blockDim.x + threadIdx.x;
  if (idx >= B_N*NP) return;
  int col = idx % NP, b = idx / NP;
  int gate = col/HP, j = col%HP;
  float acc = 0.f;
  if (j < H_N){
    const float* pr = projP + b*HP;
    const float* w  = Wi + gate*H_N + j;
    #pragma unroll 4
    for (int k=0;k<H_N;k++) acc += pr[k]*w[k*(4*H_N)];
    acc += bl[gate*H_N + j];
  }
  xcP[idx] = acc;
}

// Persistent rollout: wave = 16 batch rows, 4 waves/WG, all 60 steps on-chip.
__global__ __launch_bounds__(128) void sim_kernel(
    const float* __restrict__ projP, const float* __restrict__ xcP,
    _Float16* whPack, const float* __restrict__ wiRowP,
    const float* __restrict__ waP, const float* __restrict__ idm_params,
    const float* __restrict__ idm_s, const float* __restrict__ sdv_acts,
    const float* __restrict__ ba, float* __restrict__ out)
{
  extern __shared__ char smem[];
  _Float16* whLds  = (_Float16*)smem;                 // 114688 B (fragment-packed)
  float*    wiRowLds = (float*)(smem + 114688);       // 1792 B
  float*    waLds    = (float*)(smem + 116480);       // 448 B
  const int wave = threadIdx.x >> 5;
  const int lane = threadIdx.x & 31;
  char* wbase = smem + 116928 + wave*8192;            // per-wave: 2 x h buffer
  _Float16* hBuf0 = (_Float16*)wbase;                 // 16 x 128 f16
  _Float16* hBuf1 = (_Float16*)(wbase + 4096);

  // ---- cooperative LDS fill: Wh fragments via ASYNC global->LDS DMA if available ----
#if defined(__HIP_DEVICE_COMPILE__) && __has_builtin(__builtin_amdgcn_global_load_async_to_lds_b128)
  {
    typedef __attribute__((ext_vector_type(4))) int int4v;
    typedef __attribute__((address_space(1))) int4v g_int4;
    typedef __attribute__((address_space(3))) int4v l_int4;
    g_int4* gsrc = (g_int4*)whPack;      // 16B units
    l_int4* ldst = (l_int4*)smem;
    for (int i = threadIdx.x; i < WH_PACK_HALVES/8; i += 128)
      __builtin_amdgcn_global_load_async_to_lds_b128(gsrc + i, ldst + i, 0, 0);
  #if __has_builtin(__builtin_amdgcn_s_wait_asynccnt)
    __builtin_amdgcn_s_wait_asynccnt(0);
  #else
    asm volatile("s_wait_asynccnt 0x0" ::: "memory");
  #endif
  }
#else
  {
    const uint4* src = (const uint4*)whPack;
    uint4* dst = (uint4*)whLds;
    for (int i = threadIdx.x; i < WH_PACK_HALVES/8; i += 128) dst[i] = src[i];
  }
#endif
  for (int i = threadIdx.x; i < NP; i += 128) wiRowLds[i] = wiRowP[i];
  for (int i = threadIdx.x; i < HP; i += 128) waLds[i] = waP[i];
  __syncthreads();

  const int b0 = blockIdx.x*64 + wave*16;
  const int m  = lane & 15;
  const int hi = lane >> 4;

  // h0 = proj (f16, padded cols zero); clear second buffer
  for (int i = lane; i < 16*KP; i += 32){
    int r = i >> 7, j = i & 127;
    hBuf0[i] = (j < HP) ? (_Float16)projP[(b0+r)*HP + j] : (_Float16)0.f;
    hBuf1[i] = (_Float16)0.f;
  }

  // c0 = proj, kept in WMMA C-layout registers: element (vgpr v) = row v+8*hi, col jt*16+m
  float c[7][8];
  #pragma unroll
  for (int jt=0;jt<7;jt++)
    #pragma unroll
    for (int v=0;v<8;v++)
      c[jt][v] = projP[(b0 + v + 8*hi)*HP + jt*16 + m];

  // per-row IDM params (row b0+m; duplicated across wave halves, benign)
  const float* p = idm_params + (size_t)(b0+m)*5;
  const float desired_v = p[0], tgap = p[1], jamx = p[2], amax = p[3], amin = p[4];
  const float ba0 = ba[0];
  float ego_v = 0.f, ego_x = 0.f, prev_act = 0.f;

  const float* xcRow = xcP + (size_t)b0*NP;
  float* outActs = out;
  float* outAtts = out + (size_t)B_N*T_N;

  _Float16* hCur  = hBuf0;
  _Float16* hNext = hBuf1;

  for (int t = 0; t < T_N; ++t){
    // ---------- elementwise ego / IDM ----------
    const float* s = idm_s + ((size_t)(b0+m)*T_N + t)*12;
    if (t + 1 < T_N) __builtin_prefetch(s + 12, 0, 0);   // next step's row -> global_prefetch
    float s0=s[0], s1=s[1], s2=s[2], s3=s[3], s4=s[4], s5=s[5],
          s6=s[6], s7=s[7], s8=s[8], s9=s[9], fex=s[10], mex=s[11];
    if (t == 0){ ego_v = s0; ego_x = s3; }
    else { ego_v += prev_act*0.1f; ego_x += ego_v*0.1f + prev_act*0.005f; }
    float ef_dx = (s4 - ego_x)*fex + (1.f-fex)*s7;
    float em_dx = (s5 - ego_x)*mex + (1.f-mex)*s9;
    float ef_dv = (ego_v - s1)*fex + (1.f-fex)*s6;
    float em_dv = (ego_v - s2)*mex + (1.f-mex)*s8;
    float nf = 0.5f*hash_norm(2u*(uint32_t)t + 0u, (uint32_t)(b0+m));
    float nm = 0.5f*hash_norm(2u*(uint32_t)t + 1u, (uint32_t)(b0+m));
    float ef_act = fex*idm_act(ego_v, ef_dv, ef_dx, desired_v,tgap,jamx,amax,amin) + (1.f-fex)*nf;
    float em_act = mex*idm_act(ego_v, em_dv, em_dx, desired_v,tgap,jamx,amax,amin) + (1.f-mex)*nm;

    // sdv contribution rows for this lane's C fragment
    float sdvv[8];
    #pragma unroll
    for (int v=0;v<8;v++) sdvv[v] = sdv_acts[(size_t)(b0 + v + 8*hi)*T_N + t];

    // ---------- A fragments (h tile, 16x128 f16) from LDS ----------
    U16h afrag[4];
    #pragma unroll
    for (int kc=0;kc<4;kc++){
      int kb = kc*32;
      afrag[kc].h[0] = *(const v8h*)(hCur + m*KP + kb + hi*16);
      afrag[kc].h[1] = *(const v8h*)(hCur + m*KP + kb + hi*16 + 8);
    }

    // ---------- z = xc + sdv*WiRow + h@Wh ; gates ; c,h update ----------
    #pragma unroll 1
    for (int jt=0;jt<7;jt++){
      U8f acc[4];
      #pragma unroll
      for (int g=0;g<4;g++){
        int col = g*HP + jt*16 + m;
        float wr = wiRowLds[col];
        #pragma unroll
        for (int v=0;v<8;v++)
          acc[g].f[v] = xcRow[(v + 8*hi)*NP + col] + sdvv[v]*wr;   // L2-resident
      }
      #pragma unroll
      for (int kc=0;kc<4;kc++){
        #pragma unroll
        for (int g=0;g<4;g++){
          int f = (g*7 + jt)*4 + kc;
          v16h bfrag = *(const v16h*)(whLds + (f*32 + lane)*16);
          acc[g].v = __builtin_amdgcn_wmma_f32_16x16x32_f16(
              false, afrag[kc].v, false, bfrag, (short)0, acc[g].v, false, false);
        }
      }
      #pragma unroll
      for (int v=0;v<8;v++){
        float zi = acc[0].f[v], zf = acc[1].f[v], zg = acc[2].f[v], zo = acc[3].f[v];
        float cc = sigf(zf)*c[jt][v] + sigf(zi)*tanhf(zg);
        c[jt][v] = cc;
        float hv = sigf(zo)*tanhf(cc);
        hNext[(v + 8*hi)*KP + jt*16 + m] = (_Float16)hv;
      }
    }

    // ---------- attention + action ----------
    float dot = 0.f;
    #pragma unroll 4
    for (int j=0;j<HP;j++) dot += (float)hNext[m*KP + j]*waLds[j];   // waLds 0-padded
    float att = sigf(5.0f*(dot + ba0));
    float act = (1.f-att)*ef_act + att*em_act;
    prev_act = act;
    outActs[(size_t)(b0+m)*T_N + t] = act;
    outAtts[(size_t)(b0+m)*T_N + t] = att;

    _Float16* tmp = hCur; hCur = hNext; hNext = tmp;   // ping-pong h
  }
}

extern "C" void kernel_launch(void* const* d_in, const int* in_sizes, int n_in,
                              void* d_out, int out_size, void* d_ws, size_t ws_size,
                              hipStream_t stream){
  (void)in_sizes; (void)n_in; (void)out_size; (void)ws_size;
  const float* z   = (const float*)d_in[0];
  const float* idp = (const float*)d_in[1];
  const float* ids = (const float*)d_in[2];
  const float* sdv = (const float*)d_in[3];
  const float* W1  = (const float*)d_in[4];
  const float* b1  = (const float*)d_in[5];
  const float* Wi  = (const float*)d_in[6];
  const float* Wh  = (const float*)d_in[7];
  const float* bl  = (const float*)d_in[8];
  const float* Wa  = (const float*)d_in[9];
  const float* ba  = (const float*)d_in[10];
  float* out = (float*)d_out;

  char* ws = (char*)d_ws;
  float*    projP  = (float*)ws;     ws += (size_t)B_N*HP*sizeof(float);   // 7.34 MB
  float*    xcP    = (float*)ws;     ws += (size_t)B_N*NP*sizeof(float);   // 29.4 MB
  _Float16* whPack = (_Float16*)ws;  ws += (size_t)WH_PACK_HALVES*2;       // 114 KB
  float*    wiRowP = (float*)ws;     ws += NP*sizeof(float);
  float*    waP    = (float*)ws;     ws += HP*sizeof(float);

  proj_kernel   <<<(B_N*HP + 255)/256, 256, 0, stream>>>(z, W1, b1, projP);
  pack_wh_kernel<<<(WH_PACK_HALVES + 255)/256, 256, 0, stream>>>(Wh, whPack);
  pack_vec_kernel<<<(NP + HP + 255)/256, 256, 0, stream>>>(Wi, Wa, wiRowP, waP);
  xc_kernel     <<<(B_N*NP + 255)/256, 256, 0, stream>>>(projP, Wi, bl, xcP);

  const size_t smem = 116928 + 4*8192;   // Wh frags + vecs + 4 waves x 2 h-buffers
  sim_kernel<<<B_N/64, 128, smem, stream>>>(projP, xcP, whPack, wiRowP, waP,
                                            idp, ids, sdv, ba, out);
}
